// Attention_49237505081497
// MI455X (gfx1250) — compile-verified, matrix operand order checked by reading, not run on hardware
//
#include <hip/hip_runtime.h>

#define B_  2
#define L_  2048
#define D_  1024
#define H_  16
#define HD_ 64
#define BL_ (B_ * L_)
#define D3_ (3 * D_)

#define SROW 2052                 // padded f32 row stride for score stripe (2052 % 64 == 4)
#define QST  72                   // padded ushort row stride for Q tile
#define VCH  128                  // v-columns staged per P@V iteration
#define VST  132                  // padded ushort row stride for V^T tile
#define GST  36                   // padded ushort row stride for gemm tiles
#define SMEM_BYTES (32 * SROW * 4)

typedef __attribute__((ext_vector_type(16))) __bf16 v16bf;
typedef __attribute__((ext_vector_type(8)))  float  v8f;

union FragU { v16bf v; unsigned int u[8]; };

__device__ __forceinline__ unsigned short f2bf(float f) {
  union { __bf16 b; unsigned short u; } x;
  x.b = (__bf16)f;                       // RNE; backend picks v_cvt_*bf16
  return x.u;
}

__device__ __forceinline__ unsigned int pk2(float lo, float hi) {
  union { __bf16 b[2]; unsigned int u; } x;
  x.b[0] = (__bf16)lo;
  x.b[1] = (__bf16)hi;                   // lowers to v_cvt_pk_bf16_f32
  return x.u;
}

// 16x16x32 bf16 fragment from a bf16 row in LDS. rowp -> (row, k0).
// ISA layout: lanes 0-15: K pairs {0..7}(v0-3),{16..23}(v4-7); lanes 16-31: {8..15},{24..31}.
__device__ __forceinline__ v16bf load_frag_bf(const unsigned short* rowp, int lane) {
  const int hi = (lane >> 4) & 1;
  const unsigned int* q = (const unsigned int*)(rowp + 8 * hi);
  FragU f;
#pragma unroll
  for (int j = 0; j < 4; ++j) {
    f.u[j]     = q[j];
    f.u[j + 4] = q[j + 8];
  }
  return f.v;
}

// Same fragment, built from an f32 row (global or LDS), converting to bf16 in registers.
__device__ __forceinline__ v16bf load_frag_f32(const float* rowp, int lane) {
  const int hi = (lane >> 4) & 1;
  const float* p = rowp + 8 * hi;
  FragU f;
#pragma unroll
  for (int j = 0; j < 4; ++j) {
    f.u[j]     = pk2(p[2 * j],      p[2 * j + 1]);
    f.u[j + 4] = pk2(p[16 + 2 * j], p[16 + 2 * j + 1]);
  }
  return f.v;
}

// ------------- generic GEMM: C = A[M,K] @ B[K,N] (+bias), 64x64 tiles ---------
// 256 threads = 8 waves; wave w -> (m_off = (w&3)*16, n_off = (w>>2)*16 (+32)).
__global__ void __launch_bounds__(256)
gemm_bf16_kernel(const float* __restrict__ A, const float* __restrict__ Bm,
                 const float* __restrict__ bias, float* __restrict__ C,
                 int K, int lda, int ldb, int ldc) {
  __shared__ alignas(16) unsigned short As[64 * GST];   // [m][k]
  __shared__ alignas(16) unsigned short Bs[64 * GST];   // [n][k] (transposed fill)
  const int tid  = threadIdx.x;
  const int lane = tid & 31;
  const int wave = tid >> 5;
  const int ln   = lane & 15;
  const int hi   = lane >> 4;
  const int m0 = blockIdx.y * 64;
  const int n0 = blockIdx.x * 64;
  const int m_off = (wave & 3) * 16;
  const int n_off = (wave >> 2) * 16;
  v8f acc0 = {}, acc1 = {};
  for (int kk = 0; kk < K; kk += 32) {
#pragma unroll
    for (int i = 0; i < 4; ++i) {            // A: 64x32 -> 1024 bf16 pairs, dword stores
      int e = tid + i * 256;
      int r = e >> 4, c2 = (e & 15) * 2;
      const float* ap = &A[(size_t)(m0 + r) * lda + kk + c2];
      *(unsigned int*)&As[r * GST + c2] = pk2(ap[0], ap[1]);
    }
#pragma unroll
    for (int i = 0; i < 8; ++i) {            // B: 32k x 64n, transposed into [n][k]
      int e = tid + i * 256;
      int r = e >> 6, c = e & 63;
      Bs[c * GST + r] = f2bf(Bm[(size_t)(kk + r) * ldb + n0 + c]);
    }
    __syncthreads();
    v16bf af = load_frag_bf(&As[(m_off + ln) * GST], lane);
    v16bf b0 = load_frag_bf(&Bs[(n_off + ln) * GST], lane);
    v16bf b1 = load_frag_bf(&Bs[(n_off + 32 + ln) * GST], lane);
    acc0 = __builtin_amdgcn_wmma_f32_16x16x32_bf16(false, af, false, b0,
                                                   (short)0, acc0, false, false);
    acc1 = __builtin_amdgcn_wmma_f32_16x16x32_bf16(false, af, false, b1,
                                                   (short)0, acc1, false, false);
    __syncthreads();
  }
  const int n = n0 + n_off + ln;
  const float bv0 = bias ? bias[n]      : 0.0f;
  const float bv1 = bias ? bias[n + 32] : 0.0f;
#pragma unroll
  for (int j = 0; j < 8; ++j) {
    int m = m0 + m_off + j + 8 * hi;
    C[(size_t)m * ldc + n]      = acc0[j] + bv0;
    C[(size_t)m * ldc + n + 32] = acc1[j] + bv1;
  }
}

// ---------------- fused attention: scores -> softmax -> attn -> P@V -----------
// One block per (b,h,q-tile of 32). 256 threads = 8 waves.
// Score stripe S[32][2048] lives in LDS; attn touches HBM exactly once (the output write).
__global__ void __launch_bounds__(256)
attn_fused_kernel(const float* __restrict__ qkv, const float* __restrict__ mask,
                  float* __restrict__ attn, float* __restrict__ attn_out) {
  extern __shared__ float S[];                           // [32][SROW] f32
  __shared__ alignas(16) unsigned short Qs[32 * QST];    // Q tile, bf16
  __shared__ alignas(16) unsigned short Vst[64 * VST];   // V^T tile [d][v], bf16
  __shared__ float Mq[32];
  __shared__ float Inv[32];                              // per-row 1/sum (folded into P@V)

  const int tid  = threadIdx.x;
  const int lane = tid & 31;
  const int wave = tid >> 5;
  const int ln   = lane & 15;
  const int hi   = lane >> 4;
  const int bh = blockIdx.y;
  const int b  = bh / H_;
  const int h  = bh % H_;
  const int q0 = blockIdx.x * 32;
  const size_t base = (size_t)(b * L_) * D3_ + h * HD_;

  // Q tile -> LDS (bf16), mask row values -> LDS
#pragma unroll
  for (int i = 0; i < 8; ++i) {
    int e = tid + i * 256;
    int r = e >> 6, c = e & 63;
    Qs[r * QST + c] = f2bf(qkv[base + (size_t)(q0 + r) * D3_ + c]);
  }
  if (tid < 32) Mq[tid] = mask[b * L_ + q0 + tid];
  __syncthreads();

  // ---- phase 1: scores. Wave w owns k-columns [w*256, w*256+256). ----
  v16bf a00 = load_frag_bf(&Qs[ln * QST + 0],         lane);
  v16bf a01 = load_frag_bf(&Qs[ln * QST + 32],        lane);
  v16bf a10 = load_frag_bf(&Qs[(16 + ln) * QST + 0],  lane);
  v16bf a11 = load_frag_bf(&Qs[(16 + ln) * QST + 32], lane);

  for (int t = 0; t < 16; ++t) {
    const int k0 = wave * 256 + t * 16;
    // B fragments straight from global: K rows are contiguous over head-dim.
    const float* krow = qkv + base + (size_t)(k0 + ln) * D3_ + D_;
    v16bf b0 = load_frag_f32(krow + 0,  lane);
    v16bf b1 = load_frag_f32(krow + 32, lane);
    v8f acc0 = {}, acc1 = {};
    acc0 = __builtin_amdgcn_wmma_f32_16x16x32_bf16(false, a00, false, b0, (short)0, acc0, false, false);
    acc0 = __builtin_amdgcn_wmma_f32_16x16x32_bf16(false, a01, false, b1, (short)0, acc0, false, false);
    acc1 = __builtin_amdgcn_wmma_f32_16x16x32_bf16(false, a10, false, b0, (short)0, acc1, false, false);
    acc1 = __builtin_amdgcn_wmma_f32_16x16x32_bf16(false, a11, false, b1, (short)0, acc1, false, false);
    const int kcol = k0 + ln;
    const float mk = mask[b * L_ + kcol];
#pragma unroll
    for (int j = 0; j < 8; ++j) {
      int r0 = j + 8 * hi;
      float s0 = acc0[j]; if (Mq[r0] * mk == 0.0f)      s0 = -100000.0f;
      float s1 = acc1[j]; if (Mq[16 + r0] * mk == 0.0f) s1 = -100000.0f;
      S[r0 * SROW + kcol]        = s0;
      S[(16 + r0) * SROW + kcol] = s1;
    }
  }
  __syncthreads();

  // ---- phase 2: softmax (scale 1/sqrt(D)=1/32). Row register-resident. ----
  // S keeps unnormalized e (for P@V, normalization folded into acc); attn gets exact
  // softmax via coalesced b64 stores. float2 index = lane + 32*i.
  for (int rr = 0; rr < 4; ++rr) {
    const int row = wave * 4 + rr;
    float2* srow2 = (float2*)(S + row * SROW);
    float2 v[32];
    float m = -3.4e38f;
#pragma unroll
    for (int i = 0; i < 32; ++i) {
      v[i] = srow2[lane + i * 32];
      m = fmaxf(m, fmaxf(v[i].x, v[i].y));
    }
#pragma unroll
    for (int off = 16; off > 0; off >>= 1) m = fmaxf(m, __shfl_xor(m, off, 32));
    float sum = 0.0f;
#pragma unroll
    for (int i = 0; i < 32; ++i) {
      v[i].x = __expf((v[i].x - m) * 0.03125f);
      v[i].y = __expf((v[i].y - m) * 0.03125f);
      sum += v[i].x + v[i].y;
    }
#pragma unroll
    for (int off = 16; off > 0; off >>= 1) sum += __shfl_xor(sum, off, 32);
    const float inv = 1.0f / sum;
    if (lane == 0) Inv[row] = inv;
    float2* arow2 = (float2*)(attn + ((size_t)bh * L_ + q0 + row) * L_);
#pragma unroll
    for (int i = 0; i < 32; ++i) {
      srow2[lane + i * 32] = v[i];                      // e, for P@V
      float2 t; t.x = v[i].x * inv; t.y = v[i].y * inv;
      arow2[lane + i * 32] = t;                         // the single HBM touch of attn
    }
  }
  __syncthreads();

  // ---- phase 3: P @ V. Wave w -> sub-tile (m_off, n_off); 128-wide V staging ----
  const int m_off = (wave & 1) * 16;
  const int n_off = (wave >> 1) * 16;         // 0,16,32,48
  v8f acc = {};
  for (int v0 = 0; v0 < L_; v0 += VCH) {
#pragma unroll
    for (int i = 0; i < 32; ++i) {            // 64d x 128v = 8192 elems / 256 thr
      int e = tid + i * 256;
      int r = e >> 6, c = e & 63;             // r: v-local, c: d
      Vst[c * VST + r] = f2bf(qkv[base + (size_t)(v0 + r) * D3_ + 2 * D_ + c]);
    }
    if (v0 + VCH < L_)                        // cover latency of next chunk
      __builtin_prefetch(qkv + base + (size_t)(v0 + VCH + (tid >> 3)) * D3_ + 2 * D_, 0, 1);
    __syncthreads();
#pragma unroll
    for (int u = 0; u < 4; ++u) {             // 4 WMMAs per barrier
      v16bf af = load_frag_f32(&S[(m_off + ln) * SROW + v0 + u * 32], lane);
      v16bf bf = load_frag_bf(&Vst[(n_off + ln) * VST + u * 32], lane);
      acc = __builtin_amdgcn_wmma_f32_16x16x32_bf16(false, af, false, bf,
                                                    (short)0, acc, false, false);
    }
    __syncthreads();
  }
#pragma unroll
  for (int j = 0; j < 8; ++j) {
    int q = q0 + m_off + j + 8 * hi;
    attn_out[(size_t)(b * L_ + q) * D_ + h * HD_ + n_off + ln] =
        acc[j] * Inv[m_off + j + 8 * hi];     // fold 1/sum here
  }
}

extern "C" void kernel_launch(void* const* d_in, const int* in_sizes, int n_in,
                              void* d_out, int out_size, void* d_ws, size_t ws_size,
                              hipStream_t stream) {
  (void)in_sizes; (void)n_in; (void)out_size; (void)ws_size;
  const float* x     = (const float*)d_in[0];
  const float* mask  = (const float*)d_in[1];
  const float* W_qkv = (const float*)d_in[2];
  const float* W_h   = (const float*)d_in[3];
  const float* b_h   = (const float*)d_in[4];

  float* out  = (float*)d_out;                       // [B,L,D]
  float* attn = out + (size_t)BL_ * D_;              // [B,H,L,L]
  float* qkv      = (float*)d_ws;                    // [BL, 3D]
  float* attn_out = qkv + (size_t)BL_ * D3_;         // [BL, D]

  // Allow 256KB+ of dynamic LDS (CDNA5 WGP: 320KB per workgroup).
  hipFuncSetAttribute((const void*)attn_fused_kernel,
                      hipFuncAttributeMaxDynamicSharedMemorySize, SMEM_BYTES);

  // 1) qkv = x @ W_qkv
  gemm_bf16_kernel<<<dim3(D3_ / 64, BL_ / 64), 256, 0, stream>>>(
      x, W_qkv, nullptr, qkv, D_, D_, D3_, D3_);
  // 2) fused: scores -> softmax -> attn (single HBM touch) -> P@V
  attn_fused_kernel<<<dim3(L_ / 32, B_ * H_), 256, SMEM_BYTES, stream>>>(
      qkv, mask, attn, attn_out);
  // 3) out = attn_out @ W_h + b_h
  gemm_bf16_kernel<<<dim3(D_ / 64, BL_ / 64), 256, 0, stream>>>(
      attn_out, W_h, b_h, out, D_, D_, D_, D_);
}
